// FusedMoEModularKernel_39479339385280
// MI455X (gfx1250) — matrix operand order repaired and checked by reading, not compiled, not run on hardware
//
#include <hip/hip_runtime.h>
#include <hip/hip_bf16.h>

// Fused MoE forward for MI455X (gfx1250, wave32, WMMA + async-to-LDS).
// Pipeline: route/sort -> x:fp32->bf16 -> grouped GEMM1 (bf16 WMMA, double-
// buffered LDS; A tiles via global_load_async_to_lds_b128, weights converted
// fp32->bf16 in VALU which co-executes with WMMA) + fused silu_and_mul ->
// grouped GEMM2 (same structure) -> weighted finalize.
// The k-loop is explicitly unrolled by 2 so both LDS buffer selections are
// compile-time constants (no v_cndmask address selection on the async path).
// Workspace: 64KB header + 4MB xb + 23.1MB act(bf16) + 16.8MB y(fp32) ~ 44MB.

#define MOE_M    2048
#define MOE_K    1024
#define MOE_E    8
#define MOE_DFF  2816
#define MOE_TOPK 2
#define MOE_SLOTS (MOE_M * MOE_TOPK)   // 4096

#define BM  128
#define BN  64
#define BK  32
#define LDA 40   // LDS row stride in halves (80 B: 16B-aligned, bank-skewed)

typedef __attribute__((ext_vector_type(8)))  float  v8f;
typedef __attribute__((ext_vector_type(16))) __bf16 v16bf;
typedef __attribute__((ext_vector_type(8)))  __bf16 v8bf;

__device__ __forceinline__ unsigned short f2bf(float f) {
  union { float f; unsigned u; } v; v.f = f;
  unsigned r = v.u + 0x7FFFu + ((v.u >> 16) & 1u);   // round-to-nearest-even
  return (unsigned short)(r >> 16);
}

__device__ __forceinline__ unsigned long long pack4bf(float4 v) {
  return (unsigned long long)f2bf(v.x)
       | ((unsigned long long)f2bf(v.y) << 16)
       | ((unsigned long long)f2bf(v.z) << 32)
       | ((unsigned long long)f2bf(v.w) << 48);
}

// Async copy of 16 bytes global -> LDS, tracked by ASYNCcnt (ISA 10.7/15.18).
__device__ __forceinline__ void async_b128(unsigned lds_byte_addr,
                                           const void* gaddr) {
  asm volatile("global_load_async_to_lds_b128 %0, %1, off"
               :: "v"(lds_byte_addr), "v"(gaddr) : "memory");
}
__device__ __forceinline__ void wait_async0() {
  asm volatile("s_wait_asynccnt 0x0" ::: "memory");
}

// A-operand 16x32 bf16 fragment (ISA 7.12.2): lanes 0-15 hold K{0..7,16..23},
// lanes 16-31 hold K{8..15,24..31}.  'base' = LDS row start for this lane's M.
__device__ __forceinline__ v16bf ldfragA(const unsigned short* base, int lane) {
  const int h = (lane >> 4) * 8;
  v8bf lo = *(const v8bf*)(base + h);
  v8bf hi = *(const v8bf*)(base + 16 + h);
  return __builtin_shufflevector(lo, hi, 0,1,2,3,4,5,6,7,8,9,10,11,12,13,14,15);
}

// B-operand 32x16 bf16 fragment: lanes 0-15 hold K{0..7,16..23}, lanes 16-31 K{8..15,24..31}.
__device__ __forceinline__ v16bf ldfragB(const unsigned short* base, int lane) {
  const int h = (lane >> 4) * 8;
  v8bf lo = *(const v8bf*)(base + h);
  v8bf hi = *(const v8bf*)(base + 16 + h);
  return __builtin_shufflevector(lo, hi, 0,1,2,3,4,5,6,7,8,9,10,11,12,13,14,15);
}

// ---------------- routing ----------------
__global__ void moe_route_init(int* cnt, int* cur) {
  if (threadIdx.x < MOE_E) { cnt[threadIdx.x] = 0; cur[threadIdx.x] = 0; }
}

__global__ void moe_route_count(const int* __restrict__ ids, int* __restrict__ cnt) {
  int i = blockIdx.x * blockDim.x + threadIdx.x;
  if (i < MOE_SLOTS) atomicAdd(&cnt[ids[i]], 1);
}

__global__ void moe_route_scan(const int* __restrict__ cnt, int* __restrict__ off) {
  if (threadIdx.x == 0) {
    int s = 0;
    for (int e = 0; e < MOE_E; ++e) { off[e] = s; s += cnt[e]; }
    off[MOE_E] = s;
  }
}

__global__ void moe_route_fill(const int* __restrict__ ids, const int* __restrict__ off,
                               int* __restrict__ cur, int* __restrict__ toks,
                               int* __restrict__ slot_of) {
  int i = blockIdx.x * blockDim.x + threadIdx.x;
  if (i < MOE_SLOTS) {
    int e = ids[i];
    int p = off[e] + atomicAdd(&cur[e], 1);
    toks[p] = i >> 1;          // token index (TOPK == 2)
    slot_of[i] = p;            // inverse map -> deterministic finalize
  }
}

// ---------------- x: fp32 -> bf16 (one pass, 8 elems/thread) ----------------
__global__ void moe_conv_bf16(const float* __restrict__ in,
                              unsigned short* __restrict__ out, int n8) {
  int i = blockIdx.x * blockDim.x + threadIdx.x;
  if (i < n8) {
    const float4* p = (const float4*)(in + (size_t)i * 8);
    float4 a = p[0], b = p[1];
    unsigned long long* q = (unsigned long long*)(out + (size_t)i * 8);
    q[0] = pack4bf(a);
    q[1] = pack4bf(b);
  }
}

// ---------------- GEMM1: act = silu(x@Wg^T) * (x@Wu^T), bf16 out ----------------
__global__ __launch_bounds__(256, 1)
void moe_gemm1(const unsigned short* __restrict__ xb, const float* __restrict__ w1,
               const int* __restrict__ off, const int* __restrict__ cnt,
               const int* __restrict__ toks, unsigned short* __restrict__ act) {
  const int e = blockIdx.z, mT = blockIdx.x, nT = blockIdx.y;
  const int cntE = cnt[e];
  if (mT * BM >= cntE) return;
  const int offE = off[e];
  const float* __restrict__ w1e = w1 + (size_t)e * (2 * MOE_DFF) * MOE_K;

  __shared__ __align__(16) unsigned short As[2][BM][LDA];
  __shared__ __align__(16) unsigned short Bg[2][BN][LDA];
  __shared__ __align__(16) unsigned short Bu[2][BN][LDA];

  const int tid = threadIdx.x, lane = tid & 31, wave = tid >> 5;
  const int wm = wave >> 1, wn = wave & 1;

  // Zero A LDS once: tail rows (gr >= cntE) are never async-written.
  {
    unsigned long long* p = (unsigned long long*)&As[0][0][0];
    for (int i = tid; i < (int)(sizeof(As) / 8); i += 256) p[i] = 0ull;
  }

  // A async staging map: 128 rows x 4 b128 chunks = 512, 2 per thread.
  const unsigned short* agp[2]; unsigned alds[2][2]; bool aval[2];
#pragma unroll
  for (int j = 0; j < 2; ++j) {
    int idx = tid + j * 256;
    int r = idx >> 2, ch = idx & 3;
    int gr = mT * BM + r;
    aval[j] = (gr < cntE);
    agp[j] = aval[j] ? (xb + (size_t)toks[offE + gr] * MOE_K + ch * 8) : xb;
    alds[j][0] = (unsigned)(uintptr_t)&As[0][r][ch * 8];
    alds[j][1] = (unsigned)(uintptr_t)&As[1][r][ch * 8];
  }
  // B staging: j=0..1 gate rows, j=2..3 up rows (64 rows x 8 float4 each).
  const float* bptr[4]; int br[4], bc[4];
#pragma unroll
  for (int j = 0; j < 4; ++j) {
    int idx = tid + (j & 1) * 256;
    br[j] = idx >> 3; bc[j] = (idx & 7) * 4;
    int nrow = nT * BN + br[j] + ((j >= 2) ? MOE_DFF : 0);
    bptr[j] = w1e + (size_t)nrow * MOE_K + bc[j];
  }

  v8f cg[2][2] = {}; v8f cu[2][2] = {};

  auto stage = [&](int buf, int k0) {   // buf is a literal at every call site
#pragma unroll
    for (int j = 0; j < 2; ++j)
      if (aval[j]) async_b128(alds[j][buf], agp[j] + k0);
#pragma unroll
    for (int j = 0; j < 4; ++j) {
      float4 v = *(const float4*)(bptr[j] + k0);
      unsigned long long pk = pack4bf(v);
      if (j < 2) *(unsigned long long*)&Bg[buf][br[j]][bc[j]] = pk;
      else       *(unsigned long long*)&Bu[buf][br[j]][bc[j]] = pk;
    }
  };

  auto compute = [&](int buf) {         // buf is a literal at every call site
    v16bf a[2], bgf[2], buf_[2];
#pragma unroll
    for (int mt = 0; mt < 2; ++mt)
      a[mt] = ldfragA(&As[buf][wm * 32 + mt * 16 + (lane & 15)][0], lane);
#pragma unroll
    for (int nt = 0; nt < 2; ++nt) {
      bgf[nt]  = ldfragB(&Bg[buf][wn * 32 + nt * 16 + (lane & 15)][0], lane);
      buf_[nt] = ldfragB(&Bu[buf][wn * 32 + nt * 16 + (lane & 15)][0], lane);
    }
#pragma unroll
    for (int mt = 0; mt < 2; ++mt)
#pragma unroll
      for (int nt = 0; nt < 2; ++nt) {
        cg[mt][nt] = __builtin_amdgcn_wmma_f32_16x16x32_bf16(
            false, a[mt], false, bgf[nt], (short)0, cg[mt][nt], false, false);
        cu[mt][nt] = __builtin_amdgcn_wmma_f32_16x16x32_bf16(
            false, a[mt], false, buf_[nt], (short)0, cu[mt][nt], false, false);
      }
  };

  __syncthreads();          // zero-fill visible before first async writes
  stage(0, 0);

  // Two-phase software pipeline: MOE_K/BK = 32 steps (even).
  for (int k0 = 0; k0 < MOE_K; k0 += 2 * BK) {
    wait_async0();          // own async writes for buffer 0 complete
    __syncthreads();        // publish buf0; prior readers of buf1 are done
    stage(1, k0 + BK);      // always in range (last: 992 < 1024)
    compute(0);

    wait_async0();
    __syncthreads();
    if (k0 + 2 * BK < MOE_K) stage(0, k0 + 2 * BK);
    compute(1);
  }

  // fused silu_and_mul epilogue; C/D layout: row = vgpr + (lane>=16 ? 8 : 0)
#pragma unroll
  for (int mt = 0; mt < 2; ++mt)
#pragma unroll
    for (int nt = 0; nt < 2; ++nt)
#pragma unroll
      for (int j = 0; j < 8; ++j) {
        int rT = wm * 32 + mt * 16 + j + ((lane < 16) ? 0 : 8);
        int gRow = mT * BM + rT;
        if (gRow < cntE) {
          int col = nT * BN + wn * 32 + nt * 16 + (lane & 15);
          float g = cg[mt][nt][j], u = cu[mt][nt][j];
          float s = g / (1.f + __expf(-g));
          act[(size_t)(offE + gRow) * MOE_DFF + col] = f2bf(s * u);
        }
      }
}

// ---------------- GEMM2: y = act @ w2[e]^T (fp32 out, per routed slot) ----------------
__global__ __launch_bounds__(256, 1)
void moe_gemm2(const unsigned short* __restrict__ act, const float* __restrict__ w2,
               const int* __restrict__ off, const int* __restrict__ cnt,
               float* __restrict__ y) {
  const int e = blockIdx.z, mT = blockIdx.x, nT = blockIdx.y;
  const int cntE = cnt[e];
  if (mT * BM >= cntE) return;
  const int offE = off[e];
  const float* __restrict__ w2e = w2 + (size_t)e * MOE_K * MOE_DFF;

  __shared__ __align__(16) unsigned short As[2][BM][LDA];
  __shared__ __align__(16) unsigned short Bs[2][BN][LDA];

  const int tid = threadIdx.x, lane = tid & 31, wave = tid >> 5;
  const int wm = wave >> 1, wn = wave & 1;

  {
    unsigned long long* p = (unsigned long long*)&As[0][0][0];
    for (int i = tid; i < (int)(sizeof(As) / 8); i += 256) p[i] = 0ull;
  }

  // A async staging (bf16 act): 128 rows x 4 b128 chunks = 512, 2 per thread.
  const unsigned short* agp[2]; unsigned alds[2][2]; bool aval[2];
#pragma unroll
  for (int j = 0; j < 2; ++j) {
    int idx = tid + j * 256;
    int r = idx >> 2, ch = idx & 3;
    int gr = mT * BM + r;
    aval[j] = (gr < cntE);
    agp[j] = aval[j] ? (act + (size_t)(offE + gr) * MOE_DFF + ch * 8) : act;
    alds[j][0] = (unsigned)(uintptr_t)&As[0][r][ch * 8];
    alds[j][1] = (unsigned)(uintptr_t)&As[1][r][ch * 8];
  }
  // B staging (fp32 -> bf16): 64 rows x 8 float4 = 512, 2 per thread.
  const float* bptr[2]; int br[2], bc[2];
#pragma unroll
  for (int j = 0; j < 2; ++j) {
    int idx = tid + j * 256;
    br[j] = idx >> 3; bc[j] = (idx & 7) * 4;
    bptr[j] = w2e + (size_t)(nT * BN + br[j]) * MOE_DFF + bc[j];
  }

  v8f cc[2][2] = {};

  auto stage = [&](int buf, int d0) {
#pragma unroll
    for (int j = 0; j < 2; ++j)
      if (aval[j]) async_b128(alds[j][buf], agp[j] + d0);
#pragma unroll
    for (int j = 0; j < 2; ++j) {
      float4 v = *(const float4*)(bptr[j] + d0);
      *(unsigned long long*)&Bs[buf][br[j]][bc[j]] = pack4bf(v);
    }
  };

  auto compute = [&](int buf) {
    v16bf a[2], b[2];
#pragma unroll
    for (int mt = 0; mt < 2; ++mt)
      a[mt] = ldfragA(&As[buf][wm * 32 + mt * 16 + (lane & 15)][0], lane);
#pragma unroll
    for (int nt = 0; nt < 2; ++nt)
      b[nt] = ldfragB(&Bs[buf][wn * 32 + nt * 16 + (lane & 15)][0], lane);
#pragma unroll
    for (int mt = 0; mt < 2; ++mt)
#pragma unroll
      for (int nt = 0; nt < 2; ++nt)
        cc[mt][nt] = __builtin_amdgcn_wmma_f32_16x16x32_bf16(
            false, a[mt], false, b[nt], (short)0, cc[mt][nt], false, false);
  };

  __syncthreads();
  stage(0, 0);

  // Two-phase software pipeline: MOE_DFF/BK = 88 steps (even).
  for (int d0 = 0; d0 < MOE_DFF; d0 += 2 * BK) {
    wait_async0();
    __syncthreads();
    stage(1, d0 + BK);      // always in range (last: 2784 < 2816)
    compute(0);

    wait_async0();
    __syncthreads();
    if (d0 + 2 * BK < MOE_DFF) stage(0, d0 + 2 * BK);
    compute(1);
  }

#pragma unroll
  for (int mt = 0; mt < 2; ++mt)
#pragma unroll
    for (int nt = 0; nt < 2; ++nt)
#pragma unroll
      for (int j = 0; j < 8; ++j) {
        int rT = wm * 32 + mt * 16 + j + ((lane < 16) ? 0 : 8);
        int gRow = mT * BM + rT;
        if (gRow < cntE) {
          int col = nT * BN + wn * 32 + nt * 16 + (lane & 15);
          y[(size_t)(offE + gRow) * MOE_K + col] = cc[mt][nt][j];
        }
      }
}

// ---------------- finalize: out[m] = sum_t w[m,t] * y[slot(m,t)] ----------------
__global__ void moe_finalize(const float* __restrict__ y, const float* __restrict__ tw,
                             const int* __restrict__ slot_of, float* __restrict__ out) {
  int idx = blockIdx.x * blockDim.x + threadIdx.x;
  if (idx >= MOE_M * MOE_K) return;
  int m = idx >> 10;            // MOE_K = 1024
  int k = idx & (MOE_K - 1);
  float acc = tw[m * 2 + 0] * y[(size_t)slot_of[m * 2 + 0] * MOE_K + k]
            + tw[m * 2 + 1] * y[(size_t)slot_of[m * 2 + 1] * MOE_K + k];
  out[idx] = acc;
}

extern "C" void kernel_launch(void* const* d_in, const int* in_sizes, int n_in,
                              void* d_out, int out_size, void* d_ws, size_t ws_size,
                              hipStream_t stream) {
  const float* x   = (const float*)d_in[0];
  const float* w1  = (const float*)d_in[1];
  const float* w2  = (const float*)d_in[2];
  const float* tw  = (const float*)d_in[3];
  const int*   ids = (const int*)d_in[4];
  float* out = (float*)d_out;

  char* ws = (char*)d_ws;
  int* cnt     = (int*)(ws + 0);
  int* cur     = (int*)(ws + 64);
  int* off     = (int*)(ws + 128);               // E+1 ints
  int* toks    = (int*)(ws + 4096);              // MOE_SLOTS ints
  int* slot_of = (int*)(ws + 4096 + 4 * MOE_SLOTS);
  unsigned short* xb  = (unsigned short*)(ws + 65536);
  unsigned short* act = (unsigned short*)(ws + 65536 + (size_t)MOE_M * MOE_K * 2);
  float* y = (float*)(ws + 65536 + (size_t)MOE_M * MOE_K * 2
                      + (size_t)MOE_SLOTS * MOE_DFF * 2);

  moe_route_init<<<1, 32, 0, stream>>>(cnt, cur);
  moe_route_count<<<(MOE_SLOTS + 255) / 256, 256, 0, stream>>>(ids, cnt);
  moe_route_scan<<<1, 32, 0, stream>>>(cnt, off);
  moe_route_fill<<<(MOE_SLOTS + 255) / 256, 256, 0, stream>>>(ids, off, cur, toks, slot_of);

  const int n8 = MOE_M * MOE_K / 8;
  moe_conv_bf16<<<(n8 + 255) / 256, 256, 0, stream>>>(x, xb, n8);

  dim3 g1(MOE_SLOTS / BM, MOE_DFF / BN, MOE_E);  // (32, 44, 8), early-exit per expert
  moe_gemm1<<<g1, 256, 0, stream>>>(xb, w1, off, cnt, toks, act);

  dim3 g2(MOE_SLOTS / BM, MOE_K / BN, MOE_E);    // (32, 16, 8)
  moe_gemm2<<<g2, 256, 0, stream>>>(act, w2, off, cnt, y);

  moe_finalize<<<(MOE_M * MOE_K) / 256, 256, 0, stream>>>(y, tw, slot_of, out);
}
